// FeatureGraph_74955769249856
// MI455X (gfx1250) — matrix operand order repaired and checked by reading, not compiled, not run on hardware
//
#include <hip/hip_runtime.h>
#include <hip/hip_bf16.h>
#include <float.h>

typedef __attribute__((ext_vector_type(16))) _Float16 v16h;
typedef __attribute__((ext_vector_type(8)))  float    v8f;

#define BB   4
#define NN   512
#define CIN  256
#define DD   256
#define KK   20
#define ROWS (BB * NN)   // 2048

// ---------------------------------------------------------------------------
// Kernel 1: x_l = x*W_l^T + b_l ; x_rT[b,d,j] = (x*W_r^T + b_r)[b,j,d]
// One wave computes a 64x16 output tile: 4 M-subtiles share one B-fragment,
// 4 WMMAs per k-step, K=256 in 8 k-steps -> 32 WMMAs per wave.
// grid = (ROWS/64, DD/16, 2), block = 32  (EXEC all-ones, required for WMMA)
// ---------------------------------------------------------------------------
__global__ void __launch_bounds__(32)
proj_wmma(const float* __restrict__ x,
          const float* __restrict__ Wl, const float* __restrict__ bl,
          const float* __restrict__ Wr, const float* __restrict__ br,
          float* __restrict__ xl, float* __restrict__ xrt)
{
    const int lane  = threadIdx.x;          // 0..31
    const int mt    = blockIdx.x;           // 64-row tile
    const int nt    = blockIdx.y;           // 16-col output-dim tile
    const int which = blockIdx.z;           // 0 = W_l path, 1 = W_r path

    const float* __restrict__ W    = which ? Wr : Wl;
    const float* __restrict__ bias = which ? br : bl;

    const int m  = lane & 15;               // A-matrix row within 16-row subtile
    const int hi = lane >> 4;               // A-matrix K-half select
    const int n  = nt * 16 + (lane & 15);   // output column (C/D layout: N = lane&15)

    // Accumulators pre-loaded with bias (bias depends on N only -> same value).
    const float bv = bias[n];
    v8f c0, c1, c2, c3;
    #pragma unroll
    for (int r = 0; r < 8; ++r) { c0[r] = bv; c1[r] = bv; c2[r] = bv; c3[r] = bv; }

    const float* __restrict__ xrow0 = x + (mt * 64 +  0 + m) * CIN;
    const float* __restrict__ xrow1 = x + (mt * 64 + 16 + m) * CIN;
    const float* __restrict__ xrow2 = x + (mt * 64 + 32 + m) * CIN;
    const float* __restrict__ xrow3 = x + (mt * 64 + 48 + m) * CIN;

    #pragma unroll
    for (int k0 = 0; k0 < CIN; k0 += 32) {
        // B 32x16 f16: lane holds column N=lane&15; element e -> K in table order (same as A).
        // B[k][d] = W[d][k] (out[m,d] = sum_c x[m,c]*W[d,c]).
        v16h bmat;
        #pragma unroll
        for (int e = 0; e < 16; ++e) {
            bmat[e] = (_Float16)W[n * CIN + k0 + ((e < 8) ? e : (8 + e)) + hi * 8];
        }
        // A 16x32 f16 layout (ISA 7.12.2): lane (m,hi); element e -> VGPR e/2,
        // half e&1;  k = (v<4 ? 2v : 16+2(v-4)) + h + hi*8  (two contiguous
        // 8-float runs per lane -> compiler emits 2x global_load_b128).
        v16h a0, a1, a2, a3;
        #pragma unroll
        for (int e = 0; e < 16; ++e) {
            const int v = e >> 1, h = e & 1;
            const int k = k0 + ((v < 4) ? (v * 2) : (16 + (v - 4) * 2)) + h + hi * 8;
            a0[e] = (_Float16)xrow0[k];
            a1[e] = (_Float16)xrow1[k];
            a2[e] = (_Float16)xrow2[k];
            a3[e] = (_Float16)xrow3[k];
        }
        c0 = __builtin_amdgcn_wmma_f32_16x16x32_f16(false, a0, false, bmat, (short)0, c0, false, false);
        c1 = __builtin_amdgcn_wmma_f32_16x16x32_f16(false, a1, false, bmat, (short)0, c1, false, false);
        c2 = __builtin_amdgcn_wmma_f32_16x16x32_f16(false, a2, false, bmat, (short)0, c2, false, false);
        c3 = __builtin_amdgcn_wmma_f32_16x16x32_f16(false, a3, false, bmat, (short)0, c3, false, false);
    }

    // D layout: acc slot r -> (M = r + hi*8, N = lane&15)
    #pragma unroll
    for (int q = 0; q < 4; ++q) {
        const v8f& c = (q == 0) ? c0 : (q == 1) ? c1 : (q == 2) ? c2 : c3;
        #pragma unroll
        for (int r = 0; r < 8; ++r) {
            const int M = mt * 64 + q * 16 + r + hi * 8;   // global row (b*512 + j)
            if (which == 0) {
                xl[M * DD + n] = c[r];
            } else {
                const int b = M >> 9;                      // / 512
                const int j = M & 511;
                xrt[b * (DD * NN) + n * NN + j] = c[r];
            }
        }
    }
}

// ---------------------------------------------------------------------------
// Kernel 2: per row (b,i): alpha[j] = sum_d att_d * leaky(xl[i,d] + xr[j,d]),
// nan_to_num, top-20 (lowest-index tiebreak), softmax, emit edges + weights.
// Irreducibly VALU: leaky() sits between the add and the att-dot, so this is
// not expressible as a matmul. float2 loads: thread t owns columns 2t, 2t+1.
// grid = ROWS, block = 256 (8 waves)
// ---------------------------------------------------------------------------
__global__ void __launch_bounds__(256)
attn_topk(const float* __restrict__ xl,
          const float* __restrict__ xrt,
          const float* __restrict__ att,
          float* __restrict__ out)
{
    __shared__ float sl[DD];
    __shared__ float sa[DD];
    __shared__ float salpha[NN];
    __shared__ float rv[256];
    __shared__ int   ri[256];
    __shared__ float topv[KK];
    __shared__ int   topi[KK];

    const int row = blockIdx.x;          // b*N + i
    const int b   = row >> 9;
    const int t   = threadIdx.x;

    sl[t] = xl[row * DD + t];
    sa[t] = att[t];
    __syncthreads();

    // xrt is d-major: xr2[d*256 + t] = columns (2t, 2t+1) -> b64 coalesced.
    const float2* __restrict__ xr2 =
        (const float2*)(xrt + (size_t)b * (DD * NN));

    float acc0 = 0.f, acc1 = 0.f;
    #pragma unroll 8
    for (int d = 0; d < DD; ++d) {
        const float2 w = xr2[d * (NN / 2) + t];
        const float l  = sl[d];
        const float ad = sa[d];
        float z0 = l + w.x;
        float z1 = l + w.y;
        z0 = (z0 >= 0.f) ? z0 : 0.2f * z0;           // leaky_relu(0.2)
        z1 = (z1 >= 0.f) ? z1 : 0.2f * z1;
        acc0 = fmaf(ad, z0, acc0);
        acc1 = fmaf(ad, z1, acc1);
    }
    if (!__builtin_isfinite(acc0)) acc0 = 0.f;       // nan_to_num
    if (!__builtin_isfinite(acc1)) acc1 = 0.f;
    salpha[2 * t]     = acc0;
    salpha[2 * t + 1] = acc1;
    __syncthreads();

    // Iterative top-K: 20 x parallel argmax over 512 (tie -> lowest index).
    for (int k = 0; k < KK; ++k) {
        float v0 = salpha[t], v1 = salpha[t + 256];
        float bvv; int bii;
        if (v1 > v0) { bvv = v1; bii = t + 256; } else { bvv = v0; bii = t; }
        rv[t] = bvv; ri[t] = bii;
        __syncthreads();
        #pragma unroll
        for (int s = 128; s > 0; s >>= 1) {
            if (t < s) {
                const float ov = rv[t + s];
                const int   oi = ri[t + s];
                if (ov > rv[t] || (ov == rv[t] && oi < ri[t])) { rv[t] = ov; ri[t] = oi; }
            }
            __syncthreads();
        }
        if (t == 0) {
            topv[k] = rv[0];
            topi[k] = ri[0];
            salpha[ri[0]] = -FLT_MAX;    // remove winner
        }
        __syncthreads();
    }

    // Softmax over the 20 selected (already descending -> max = topv[0]).
    if (t == 0) {
        const float mx = topv[0];
        float e[KK], s = 0.f;
        #pragma unroll
        for (int k = 0; k < KK; ++k) { e[k] = __expf(topv[k] - mx); s += e[k]; }
        const float inv = 1.f / s;
        #pragma unroll
        for (int k = 0; k < KK; ++k) topv[k] = e[k] * inv;
    }
    __syncthreads();

    // Output (all float): [0,40960) index_i ; [40960,81920) index_j ; then attn
    if (t < KK) {
        const int base = row * KK + t;
        out[base]                 = (float)row;                 // i + b*N
        out[ROWS * KK + base]     = (float)(b * NN + topi[t]);  // topk idx + b*N
        out[2 * ROWS * KK + base] = topv[t];
    }
}

extern "C" void kernel_launch(void* const* d_in, const int* in_sizes, int n_in,
                              void* d_out, int out_size, void* d_ws, size_t ws_size,
                              hipStream_t stream) {
    const float* x   = (const float*)d_in[0];
    // d_in[1] = edge_index (int64, unused), d_in[2] = batch (int64, unused)
    const float* Wl  = (const float*)d_in[3];
    const float* bl  = (const float*)d_in[4];
    const float* Wr  = (const float*)d_in[5];
    const float* br  = (const float*)d_in[6];
    const float* att = (const float*)d_in[7];
    float* out = (float*)d_out;

    // workspace: x_l (2048*256 f32 = 2MB) then x_rT (4*256*512 f32 = 2MB)
    float* xl  = (float*)d_ws;
    float* xrt = xl + (size_t)ROWS * DD;

    dim3 g1(ROWS / 64, DD / 16, 2);
    proj_wmma<<<g1, 32, 0, stream>>>(x, Wl, bl, Wr, br, xl, xrt);
    attn_topk<<<ROWS, 256, 0, stream>>>(xl, xrt, att, out);
}
